// StreamModelForGemma_33973191311840
// MI455X (gfx1250) — matrix operand order repaired and checked by reading, not compiled, not run on hardware
//
#include <hip/hip_runtime.h>

// ---------------------------------------------------------------------------
// Types for CDNA5 WMMA (wave32)
// ---------------------------------------------------------------------------
typedef __attribute__((ext_vector_type(16))) __bf16 v16bf;
typedef __attribute__((ext_vector_type(8)))  __bf16 v8bf;
typedef __attribute__((ext_vector_type(8)))  float  v8f;

union v16u { v16bf v; v8bf h[2]; };
static __device__ __forceinline__ v16bf mk16(v8bf lo, v8bf hi) {
    v16u u; u.h[0] = lo; u.h[1] = hi; return u.v;
}

static __device__ __forceinline__ v8f wmma_bf16(v16bf a, v16bf b, v8f c) {
    // D = A(16x32 bf16) x B(32x16 bf16) + C(16x16 f32)
    return __builtin_amdgcn_wmma_f32_16x16x32_bf16(false, a, false, b, (short)0, c, false, false);
}

// ---------------------------------------------------------------------------
// CDNA5 async global->LDS copy (ASYNCcnt-tracked, no VGPR round trip).
// Low 32 bits of a generic pointer to __shared__ == LDS byte offset.
// ---------------------------------------------------------------------------
static __device__ __forceinline__ void async_ld_b128(void* lds, const void* g) {
    asm volatile("global_load_async_to_lds_b128 %0, %1, off"
                 :: "v"((unsigned)(unsigned long long)lds),
                    "v"((unsigned long long)g)
                 : "memory");
}
static __device__ __forceinline__ void wait_async() {
    asm volatile("s_wait_asynccnt 0x0" ::: "memory");
}

// ---------------------------------------------------------------------------
// Problem constants
// ---------------------------------------------------------------------------
#define B_    2
#define S_    4096
#define HID_  2048
#define NH_   8
#define NKV_  4
#define D_    256
#define SCALE_ 0.0625f           // D^-0.5 = 1/16
#define EPS_   1e-6f

// ---------------------------------------------------------------------------
// f32 -> bf16 conversion (grid-stride)
// ---------------------------------------------------------------------------
__global__ void f32_to_bf16(const float* __restrict__ in, __bf16* __restrict__ out, long n) {
    long i = (long)blockIdx.x * blockDim.x + threadIdx.x;
    long stride = (long)gridDim.x * blockDim.x;
    for (; i < n; i += stride) out[i] = (__bf16)in[i];
}

// ---------------------------------------------------------------------------
// Tiled f32[K][N] -> bf16[N][K] transpose (coalesced both sides via LDS).
// Block (32,8); grid (N/32, K/32).
// ---------------------------------------------------------------------------
__global__ __launch_bounds__(256) void transpose_to_bf16(const float* __restrict__ in,
                                                         __bf16* __restrict__ out,
                                                         int K, int N) {
    __shared__ float t[32][33];
    const int kb = blockIdx.y * 32, nb = blockIdx.x * 32;
    const int tx = threadIdx.x, ty = threadIdx.y;
#pragma unroll
    for (int i = 0; i < 32; i += 8)
        t[ty + i][tx] = in[(size_t)(kb + ty + i) * N + nb + tx];
    __syncthreads();
#pragma unroll
    for (int i = 0; i < 32; i += 8)
        out[(size_t)(nb + ty + i) * K + kb + tx] = (__bf16)t[tx][ty + i];
}

// ---------------------------------------------------------------------------
// Tiled bf16 WMMA GEMM:  C[M,N] = A[M,K] * BT[N,K]^T   (BT pre-transposed)
// Block tile 128x128, BK=32, 128 threads = 4 waves (2x2), 64x64 per wave,
// 4x4 WMMA accumulators. Double-buffered LDS, async global->LDS staging.
// TOUT=true writes C transposed ([N][M]) for the d-major V layout.
// ---------------------------------------------------------------------------
template <typename OutT, bool TOUT>
__global__ __launch_bounds__(128) void gemm_bf16(const __bf16* __restrict__ A,
                                                 const __bf16* __restrict__ BT,
                                                 OutT* __restrict__ C,
                                                 int M, int N, int K) {
    __shared__ __bf16 As[2][128][40];   // [m][k], pad 8 (16B-aligned rows)
    __shared__ __bf16 Bs[2][128][40];   // [n][k]

    const int tid  = threadIdx.x;
    const int w    = tid >> 5;
    const int lane = tid & 31;
    const int hf   = lane >> 4;
    const int l16  = lane & 15;
    const int wm   = w >> 1;            // 0..1
    const int wn   = w & 1;             // 0..1
    const int m0 = blockIdx.y * 128;
    const int n0 = blockIdx.x * 128;

    v8f acc[4][4] = {};

    auto stage = [&](int k0, int buf) {
#pragma unroll
        for (int i = 0; i < 4; ++i) {
            int id  = tid * 4 + i;          // 0..511
            int row = id >> 2, cc = id & 3; // 128 rows x 4 chunks of 8 bf16
            async_ld_b128(&As[buf][row][cc * 8], A  + (size_t)(m0 + row) * K + k0 + cc * 8);
        }
#pragma unroll
        for (int i = 0; i < 4; ++i) {
            int id  = tid * 4 + i;
            int row = id >> 2, cc = id & 3;
            async_ld_b128(&Bs[buf][row][cc * 8], BT + (size_t)(n0 + row) * K + k0 + cc * 8);
        }
    };

    stage(0, 0);
    wait_async();
    __syncthreads();

    int cur = 0;
    for (int k0 = 0; k0 < K; k0 += 32) {
        if (k0 + 32 < K) stage(k0 + 32, cur ^ 1);   // prefetch next tile

        v16bf afrag[4], bfrag[4];
#pragma unroll
        for (int mt = 0; mt < 4; ++mt) {
            const __bf16* ap = &As[cur][wm * 64 + mt * 16 + l16][0];
            afrag[mt] = mk16(*(const v8bf*)(ap + hf * 8), *(const v8bf*)(ap + 16 + hf * 8));
        }
#pragma unroll
        for (int nt = 0; nt < 4; ++nt) {
            const __bf16* bp = &Bs[cur][wn * 64 + nt * 16 + l16][0];
            bfrag[nt] = mk16(*(const v8bf*)(bp + hf * 8), *(const v8bf*)(bp + 16 + hf * 8));
        }
#pragma unroll
        for (int mt = 0; mt < 4; ++mt)
#pragma unroll
            for (int nt = 0; nt < 4; ++nt)
                acc[mt][nt] = wmma_bf16(afrag[mt], bfrag[nt], acc[mt][nt]);

        wait_async();
        __syncthreads();
        cur ^= 1;
    }

#pragma unroll
    for (int mt = 0; mt < 4; ++mt)
#pragma unroll
        for (int nt = 0; nt < 4; ++nt)
#pragma unroll
            for (int r = 0; r < 8; ++r) {
                int row = m0 + wm * 64 + mt * 16 + hf * 8 + r;
                int col = n0 + wn * 64 + nt * 16 + l16;
                if (TOUT) C[(size_t)col * M + row] = (OutT)acc[mt][nt][r];
                else      C[(size_t)row * N + col] = (OutT)acc[mt][nt][r];
            }
}

// ---------------------------------------------------------------------------
// Per-(token,head) RMSNorm + RoPE, one wave32 per row of D=256, in-place.
// x layout: [tok][head][D] bf16; cos/sin: [tok][D] f32 (B,S flattened).
// ---------------------------------------------------------------------------
__global__ __launch_bounds__(32) void norm_rope(__bf16* __restrict__ x,
                                                const float* __restrict__ w,
                                                const float* __restrict__ cs,
                                                const float* __restrict__ sn,
                                                int nheads) {
    const int h    = blockIdx.x;
    const long tok = blockIdx.y;
    const int lane = threadIdx.x;
    const size_t base = ((size_t)tok * nheads + h) * D_;
    const int d0  = lane * 8;
    const int dp0 = (d0 + 128) & 255;

    v8bf xv = *(const v8bf*)(x + base + d0);
    v8bf pv = *(const v8bf*)(x + base + dp0);

    float xs[8], ps[8];
    float ss = 0.f;
#pragma unroll
    for (int j = 0; j < 8; ++j) {
        xs[j] = (float)xv[j];
        ps[j] = (float)pv[j];
        ss += xs[j] * xs[j];
    }
#pragma unroll
    for (int m = 1; m < 32; m <<= 1) ss += __shfl_xor(ss, m);
    const float rinv = rsqrtf(ss * (1.0f / 256.0f) + EPS_);

    const size_t cbase = (size_t)tok * D_;
#pragma unroll
    for (int j = 0; j < 8; ++j) {
        int d = d0 + j;
        float c = cs[cbase + d];
        float s = sn[cbase + d];
        float xn = xs[j] * rinv * (1.0f + w[d]);
        float pn = ps[j] * rinv * (1.0f + w[(d + 128) & 255]);
        float rot = (d < 128) ? -pn : pn;
        x[base + d] = (__bf16)(xn * c + rot * s);
    }
}

// ---------------------------------------------------------------------------
// Flash attention (causal, GQA). 4 waves / WG; wave w owns query rows
// [qt*64 + w*16, +16). K-tiles of 32 tokens, double-buffered async staging.
// Q [tok][NH][D], K [tok][NKV][D], VT d-major [NKV*D][B*S] (all bf16).
// Output O: [tok][NH][D] bf16.
// ---------------------------------------------------------------------------
__global__ __launch_bounds__(128) void attn_kernel(const __bf16* __restrict__ Q,
                                                   const __bf16* __restrict__ Kb,
                                                   const __bf16* __restrict__ VT,
                                                   __bf16* __restrict__ O) {
    constexpr int KT = 32;
    __shared__ __bf16 lds_k[2][KT][D_ + 8];     // [ktok][d]  (B-operand of Q*K^T)
    __shared__ __bf16 lds_v[2][D_][KT + 8];     // [d][ktok]  (B-operand of P*V)
    __shared__ __bf16 lds_p[4][16][KT + 8];     // per-wave P relay (C->A layout)

    const int tid  = threadIdx.x;
    const int w    = tid >> 5;
    const int lane = tid & 31;
    const int hf   = lane >> 4;
    const int l16  = lane & 15;
    const int qt = blockIdx.x, h = blockIdx.y, b = blockIdx.z;
    const int kvh = h >> 1;                     // NH/NKV = 2
    const int qrow0 = qt * 64 + w * 16;

    // Q A-fragments for this wave's 16 rows, all of D (8 chunks of K=32)
    v16bf qf[8];
    {
        const size_t qbase = (((size_t)b * S_ + (qrow0 + l16)) * NH_ + h) * D_;
#pragma unroll
        for (int c = 0; c < 8; ++c) {
            v8bf lo = *(const v8bf*)(Q + qbase + c * 32 + hf * 8);
            v8bf hi = *(const v8bf*)(Q + qbase + c * 32 + 16 + hf * 8);
            qf[c] = mk16(lo, hi);
        }
    }

    auto stageKV = [&](int kt, int buf) {
        const int kbase = kt * KT;
#pragma unroll
        for (int i = 0; i < 8; ++i) {           // K: 32 tok x 256 d = 1024 chunks
            int id = i * 128 + tid;
            int row = id >> 5, cc = id & 31;
            async_ld_b128(&lds_k[buf][row][cc * 8],
                          Kb + (((size_t)b * S_ + kbase + row) * NKV_ + kvh) * D_ + cc * 8);
        }
#pragma unroll
        for (int i = 0; i < 8; ++i) {           // V: 256 d x 32 tok = 1024 chunks
            int id = i * 128 + tid;
            int d = id >> 2, cc = id & 3;
            async_ld_b128(&lds_v[buf][d][cc * 8],
                          VT + ((size_t)(kvh * D_ + d)) * (B_ * S_) + (size_t)b * S_ + kbase + cc * 8);
        }
    };

    v8f acc[16] = {};           // O accumulators: 16 col-tiles of 16 (D=256)
    float m[8], lsum[8];        // per-lane softmax state for rows 8*hf + r
#pragma unroll
    for (int r = 0; r < 8; ++r) { m[r] = -1e30f; lsum[r] = 0.f; }

    const int nkt = 2 * qt + 2;                 // causal: tiles up to & incl. diagonal
    stageKV(0, 0);
    wait_async();
    __syncthreads();

    int cur = 0;
    for (int kt = 0; kt < nkt; ++kt) {
        const int kbase = kt * KT;
        if (kt + 1 < nkt) stageKV(kt + 1, cur ^ 1);   // prefetch next K/V tile

        // ---- scores: two independent WMMA chains interleaved over the K dim ----
        float p[2][8];
        {
            v8f sc0 = {}, sc1 = {};
#pragma unroll
            for (int c = 0; c < 8; ++c) {
                const __bf16* bp0 = &lds_k[cur][l16][c * 32 + hf * 8];
                const __bf16* bp1 = &lds_k[cur][16 + l16][c * 32 + hf * 8];
                v16bf b0 = mk16(*(const v8bf*)bp0, *(const v8bf*)(bp0 + 16));
                v16bf b1 = mk16(*(const v8bf*)bp1, *(const v8bf*)(bp1 + 16));
                sc0 = wmma_bf16(qf[c], b0, sc0);
                sc1 = wmma_bf16(qf[c], b1, sc1);
            }
            // scale + causal mask; C layout: elem r -> (row 8*hf+r, col l16)
            float tmax[8];
#pragma unroll
            for (int r = 0; r < 8; ++r) {
                int qr = qrow0 + hf * 8 + r;
                float s0 = (kbase + l16      <= qr) ? sc0[r] * SCALE_ : -1e30f;
                float s1 = (kbase + 16 + l16 <= qr) ? sc1[r] * SCALE_ : -1e30f;
                p[0][r] = s0; p[1][r] = s1;
                tmax[r] = fmaxf(s0, s1);
            }
#pragma unroll
            for (int msk = 1; msk < 16; msk <<= 1)
#pragma unroll
                for (int r = 0; r < 8; ++r)
                    tmax[r] = fmaxf(tmax[r], __shfl_xor(tmax[r], msk));

            float alpha[8], rs[8];
#pragma unroll
            for (int r = 0; r < 8; ++r) {
                float mn = fmaxf(m[r], tmax[r]);
                alpha[r] = __expf(m[r] - mn);
                m[r] = mn;
                p[0][r] = __expf(p[0][r] - mn);
                p[1][r] = __expf(p[1][r] - mn);
                rs[r] = p[0][r] + p[1][r];
            }
#pragma unroll
            for (int msk = 1; msk < 16; msk <<= 1)
#pragma unroll
                for (int r = 0; r < 8; ++r) rs[r] += __shfl_xor(rs[r], msk);
#pragma unroll
            for (int r = 0; r < 8; ++r) lsum[r] = lsum[r] * alpha[r] + rs[r];
#pragma unroll
            for (int n = 0; n < 16; ++n)
#pragma unroll
                for (int r = 0; r < 8; ++r) acc[n][r] *= alpha[r];
            // stage P (bf16) into per-wave LDS strip, row-major [qrow][kcol]
#pragma unroll
            for (int nt = 0; nt < 2; ++nt)
#pragma unroll
                for (int r = 0; r < 8; ++r)
                    lds_p[w][hf * 8 + r][nt * 16 + l16] = (__bf16)p[nt][r];
        }

        // ---- P(16x32) x V(32x256): 16 independent WMMAs, grouped 4 loads ahead ----
        {
            const __bf16* pr = &lds_p[w][l16][0];
            v16bf af = mk16(*(const v8bf*)(pr + hf * 8), *(const v8bf*)(pr + 16 + hf * 8));
#pragma unroll
            for (int g = 0; g < 4; ++g) {
                v16bf bf[4];
#pragma unroll
                for (int j = 0; j < 4; ++j) {
                    const __bf16* vp = &lds_v[cur][(g * 4 + j) * 16 + l16][hf * 8];
                    bf[j] = mk16(*(const v8bf*)vp, *(const v8bf*)(vp + 16));
                }
#pragma unroll
                for (int j = 0; j < 4; ++j)
                    acc[g * 4 + j] = wmma_bf16(af, bf[j], acc[g * 4 + j]);
            }
        }
        wait_async();
        __syncthreads();   // next-tile async writes done; all reads of cur done
        cur ^= 1;
    }

    // ---- normalize and write O (token-major [tok][NH][D]) ----
#pragma unroll
    for (int r = 0; r < 8; ++r) {
        float inv = 1.0f / lsum[r];
        int qr = qrow0 + hf * 8 + r;
        size_t ob = (((size_t)b * S_ + qr) * NH_ + h) * D_;
#pragma unroll
        for (int n = 0; n < 16; ++n)
            O[ob + n * 16 + l16] = (__bf16)(acc[n][r] * inv);
    }
}

// ---------------------------------------------------------------------------
// Host-side launcher
// ---------------------------------------------------------------------------
extern "C" void kernel_launch(void* const* d_in, const int* in_sizes, int n_in,
                              void* d_out, int out_size, void* d_ws, size_t ws_size,
                              hipStream_t stream) {
    (void)in_sizes; (void)n_in; (void)out_size; (void)ws_size;
    const float* hs   = (const float*)d_in[0];
    const float* cosT = (const float*)d_in[1];
    const float* sinT = (const float*)d_in[2];
    const float* wq   = (const float*)d_in[3];
    const float* wk   = (const float*)d_in[4];
    const float* wv   = (const float*)d_in[5];
    const float* wo   = (const float*)d_in[6];
    const float* qnw  = (const float*)d_in[7];
    const float* knw  = (const float*)d_in[8];
    float* out = (float*)d_out;

    // workspace partition (bf16 elements), ~159 MB total
    __bf16* p    = (__bf16*)d_ws;
    __bf16* hsb  = p; p += (size_t)16777216;   // 8192 x 2048
    __bf16* wqbT = p; p += (size_t)4194304;    // 2048 x 2048 (transposed [N][K])
    __bf16* wkbT = p; p += (size_t)2097152;    // 1024 x 2048
    __bf16* wvbT = p; p += (size_t)2097152;    // 1024 x 2048
    __bf16* wobT = p; p += (size_t)4194304;    // 2048 x 2048
    __bf16* qb   = p; p += (size_t)16777216;   // 8192 x 8 x 256  (token-major)
    __bf16* kb   = p; p += (size_t)8388608;    // 8192 x 4 x 256  (token-major)
    __bf16* vbT  = p; p += (size_t)8388608;    // 1024 x 8192     (d-major)
    __bf16* ab   = p; p += (size_t)16777216;   // 8192 x 8 x 256

    // 1) convert hidden; transpose-convert weights to [N][K] bf16
    f32_to_bf16<<<2048, 256, 0, stream>>>(hs, hsb, 16777216L);
    transpose_to_bf16<<<dim3(64, 64), dim3(32, 8), 0, stream>>>(wq, wqbT, 2048, 2048);
    transpose_to_bf16<<<dim3(32, 64), dim3(32, 8), 0, stream>>>(wk, wkbT, 2048, 1024);
    transpose_to_bf16<<<dim3(32, 64), dim3(32, 8), 0, stream>>>(wv, wvbT, 2048, 1024);
    transpose_to_bf16<<<dim3(64, 64), dim3(32, 8), 0, stream>>>(wo, wobT, 2048, 2048);

    // 2) QKV projections (async-staged WMMA GEMMs); V written d-major
    gemm_bf16<__bf16, false><<<dim3(16, 64), 128, 0, stream>>>(hsb, wqbT, qb,  8192, 2048, 2048);
    gemm_bf16<__bf16, false><<<dim3(8,  64), 128, 0, stream>>>(hsb, wkbT, kb,  8192, 1024, 2048);
    gemm_bf16<__bf16, true ><<<dim3(8,  64), 128, 0, stream>>>(hsb, wvbT, vbT, 8192, 1024, 2048);

    // 3) per-head RMSNorm + RoPE (in place)
    norm_rope<<<dim3(NH_,  B_ * S_), 32, 0, stream>>>(qb, qnw, cosT, sinT, NH_);
    norm_rope<<<dim3(NKV_, B_ * S_), 32, 0, stream>>>(kb, knw, cosT, sinT, NKV_);

    // 4) causal GQA flash attention (double-buffered async K/V staging)
    attn_kernel<<<dim3(S_ / 64, NH_, B_), 128, 0, stream>>>(qb, kb, vbT, ab);

    // 5) output projection -> f32 result
    gemm_bf16<float, false><<<dim3(16, 64), 128, 0, stream>>>(ab, wobT, out, 8192, 2048, 2048);
}
